// IrrAttention_1640677507313
// MI455X (gfx1250) — compile-verified, matrix operand order checked by reading, not run on hardware
//
#include <hip/hip_runtime.h>

// ============================================================================
// IrrAttention for MI455X (gfx1250, wave32, WMMA).
// Strategy: all GEMM-like ops via v_wmma_f32_16x16x32_bf16 (bf16 A/B, f32 acc).
// Weights pre-transposed to bf16 so B-fragments load contiguous-K b128s.
// Attention probs written f32 to d_out and re-read (bf16-converted on the fly)
// by the AV pass; 192MB L2 holds the whole working set.
// ============================================================================

#define DEV __device__ __forceinline__

typedef __attribute__((ext_vector_type(16))) __bf16          v16bf;
typedef __attribute__((ext_vector_type(8)))  float           v8f;
typedef __attribute__((ext_vector_type(8)))  unsigned short  u16x8;
typedef __attribute__((ext_vector_type(4)))  float           f32x4;

union FragU { v16bf v; u16x8 h[2]; unsigned short s[16]; };

DEV unsigned short f2bf(float f) {           // round-to-nearest-even f32->bf16
  unsigned int u = __float_as_uint(f);
  u += 0x7FFFu + ((u >> 16) & 1u);
  return (unsigned short)(u >> 16);
}

DEV v8f wmma_bf16(v16bf a, v16bf b, v8f c) {
  return __builtin_amdgcn_wmma_f32_16x16x32_bf16(false, a, false, b,
                                                 (short)0, c, false, false);
}

// A-fragment (16x32, row-major, K contiguous) or B-fragment from Bt (NxK).
// Per ISA layout: lanes 0-15 hold K {kk..kk+7, kk+16..kk+23} of row lane&15;
// lanes 16-31 hold K {kk+8..kk+15, kk+24..kk+31}.
DEV v16bf load_frag_bf16(const unsigned short* base, int row0, int rstride, int kk) {
  int lane = threadIdx.x & 31;
  const unsigned short* p =
      base + (size_t)(row0 + (lane & 15)) * rstride + kk + ((lane < 16) ? 0 : 8);
  FragU u;
  u.h[0] = *(const u16x8*)p;
  u.h[1] = *(const u16x8*)(p + 16);
  return u.v;
}

// A-fragment built from f32 source (optionally sum of two sources), converted
// to bf16.  fast==true: unguarded vector loads.  Otherwise per-element bounds
// guard (rows/cols >= nv read as 0) to handle 1023-sized attention matrices.
DEV v16bf load_frag_f32(const float* a1, const float* a2, size_t stride,
                        int row0, int kk, int nv, bool fast) {
  int lane = threadIdx.x & 31;
  int r  = row0 + (lane & 15);
  int k0 = kk + ((lane < 16) ? 0 : 8);
  FragU u;
  if (fast) {
    const float* p = a1 + (size_t)r * stride + k0;
    f32x4 x0 = *(const f32x4*)p;
    f32x4 x1 = *(const f32x4*)(p + 4);
    f32x4 x2 = *(const f32x4*)(p + 16);
    f32x4 x3 = *(const f32x4*)(p + 20);
    if (a2) {
      const float* q = a2 + (size_t)r * stride + k0;
      x0 += *(const f32x4*)q;        x1 += *(const f32x4*)(q + 4);
      x2 += *(const f32x4*)(q + 16); x3 += *(const f32x4*)(q + 20);
    }
#pragma unroll
    for (int j = 0; j < 4; ++j) {
      u.s[j]      = f2bf(x0[j]);
      u.s[4 + j]  = f2bf(x1[j]);
      u.s[8 + j]  = f2bf(x2[j]);
      u.s[12 + j] = f2bf(x3[j]);
    }
  } else {
#pragma unroll
    for (int j = 0; j < 16; ++j) {
      int k = (j < 8) ? (k0 + j) : (k0 + 8 + j);   // groups {0..7},{16..23}
      float v = 0.f;
      if (r < nv && k < nv) {
        size_t idx = (size_t)r * stride + (size_t)k;
        v = a1[idx];
        if (a2) v += a2[idx];
      }
      u.s[j] = f2bf(v);
    }
  }
  return u.v;
}

// ---------------------------------------------------------------------------
// Prep: 12 weights (512x512 f32) -> transposed bf16  Wt[n][k]
// ---------------------------------------------------------------------------
struct WPtrs { const float* w[12]; };

__global__ __launch_bounds__(256) void k_prep_w(WPtrs wp, unsigned short* wt) {
  size_t idx = (size_t)blockIdx.x * 256 + threadIdx.x;
  if (idx >= (size_t)12 * 512 * 512) return;
  int widx = (int)(idx >> 18);
  int rem  = (int)(idx & 262143);
  int i = rem >> 9;                 // k (row of W)
  int j = rem & 511;                // n (col of W)
  wt[(size_t)widx * 262144 + (size_t)j * 512 + i] = f2bf(wp.w[widx][rem]);
}

// ---------------------------------------------------------------------------
// Prep: 5 padded bf16 activation matrices [4096][512]
//   0: poi   1: poi_merge+time   2: poi_merge+dist   3: time   4: dist
// (1023-row sources zero-padded at l==1023)
// ---------------------------------------------------------------------------
__global__ __launch_bounds__(256) void k_prep_x(const float* __restrict__ poi,
                                                const float* __restrict__ tiv,
                                                const float* __restrict__ dis,
                                                unsigned short* __restrict__ xp) {
  size_t idx = (size_t)blockIdx.x * 256 + threadIdx.x;
  if (idx >= (size_t)5 * 2097152) return;
  int a = (int)(idx >> 21);
  size_t rem = idx & 2097151;
  int m = (int)(rem >> 9);
  int d = (int)(rem & 511);
  int b = m >> 10, l = m & 1023;
  size_t pidx = ((size_t)(b * 1024 + l)) * 512 + d;
  size_t iidx = ((size_t)(b * 1023 + l)) * 512 + d;
  float v = 0.f;
  if (a == 0) v = poi[pidx];
  else if (l < 1023) {
    if      (a == 1) v = poi[pidx] + poi[pidx + 512] + tiv[iidx];
    else if (a == 2) v = poi[pidx] + poi[pidx + 512] + dis[iidx];
    else if (a == 3) v = tiv[iidx];
    else             v = dis[iidx];
  }
  xp[idx] = f2bf(v);
}

// ---------------------------------------------------------------------------
// Projection GEMM: A[4096][512] bf16 x Bt[512][512] -> q/k/v bf16 [32][1024][64]
// ---------------------------------------------------------------------------
__global__ __launch_bounds__(128) void k_proj(const unsigned short* __restrict__ A,
                                              const unsigned short* __restrict__ Bt,
                                              unsigned short* __restrict__ outq) {
  int wave = threadIdx.x >> 5, lane = threadIdx.x & 31;
  int m0 = blockIdx.x * 16;
  int n0 = (blockIdx.y * 4 + wave) * 16;
  v8f acc = {};
  for (int kk = 0; kk < 512; kk += 32) {
    v16bf a = load_frag_bf16(A, m0, 512, kk);
    v16bf b = load_frag_bf16(Bt, n0, 512, kk);
    acc = wmma_bf16(a, b, acc);
  }
  int n = n0 + (lane & 15);
  int h = n >> 6, hd = n & 63;
  int mo = (lane < 16) ? 0 : 8;
#pragma unroll
  for (int i = 0; i < 8; ++i) {
    int m = m0 + i + mo;
    int b_ = m >> 10, l = m & 1023;
    outq[(((size_t)(b_ * 8 + h)) * 1024 + l) * 64 + hd] = f2bf(acc[i]);
  }
}

// ---------------------------------------------------------------------------
// Transpose pv/tv/dv [32][1024][64] -> Vt [3][32][64][1024] via LDS tiles
// ---------------------------------------------------------------------------
__global__ __launch_bounds__(256) void k_transpose_v(
    const unsigned short* __restrict__ qkv, unsigned short* __restrict__ vt) {
  __shared__ unsigned short tile[16][17];
  const int vsrc[3] = {2, 5, 8};
  int t  = blockIdx.z >> 5;
  int bh = blockIdx.z & 31;
  int l0 = blockIdx.x * 16, d0 = blockIdx.y * 16;
  int tx = threadIdx.x & 15, ty = threadIdx.x >> 4;
  const unsigned short* src = qkv + ((size_t)vsrc[t] * 32 + bh) * 65536;
  tile[ty][tx] = src[(size_t)(l0 + ty) * 64 + d0 + tx];
  __syncthreads();
  unsigned short* dst = vt + ((size_t)t * 32 + bh) * 65536;
  dst[(size_t)(d0 + ty) * 1024 + l0 + tx] = tile[tx][ty];
}

// ---------------------------------------------------------------------------
// Attention: scores (QK^T/8) + mask + softmax -> f32 probs in d_out.
// One WG per 16-query block; 8 waves x 8 col-tiles x (2 WMMAs over HD=64).
// Scores staged in 64KB dynamic LDS for the row softmax.
// ---------------------------------------------------------------------------
__global__ __launch_bounds__(256) void k_attn(const unsigned short* __restrict__ Q,
                                              const unsigned short* __restrict__ K,
                                              const int* __restrict__ mask,
                                              float* __restrict__ outp, int nv) {
  extern __shared__ float smem_dyn[];
  float* ssc     = smem_dyn;            // [16][1024]
  float* sred    = smem_dyn + 16384;    // [16][16]
  float* rowstat = sred + 256;          // [16]

  int tid = threadIdx.x, wave = tid >> 5, lane = tid & 31;
  int q0 = blockIdx.x * 16, bh = blockIdx.y;
  const unsigned short* Qb = Q + (size_t)bh * 65536 + (size_t)q0 * 64;
  const unsigned short* Kb = K + (size_t)bh * 65536;
  int bidx = bh >> 3;

  for (int t = 0; t < 8; ++t) {
    int n0 = (wave * 8 + t) * 16;
    v8f acc = {};
#pragma unroll
    for (int kk = 0; kk < 64; kk += 32) {
      v16bf a = load_frag_bf16(Qb, 0, 64, kk);
      v16bf b = load_frag_bf16(Kb, n0, 64, kk);
      acc = wmma_bf16(a, b, acc);
    }
    int col = n0 + (lane & 15);
    int mo = (lane < 16) ? 0 : 8;
#pragma unroll
    for (int i = 0; i < 8; ++i) {
      int row = i + mo;
      float s = acc[i] * 0.125f;                        // 1/sqrt(HD)
      if (mask) {
        if (mask[(size_t)bidx * 1048576 + (size_t)(q0 + row) * 1024 + col] == 0)
          s = -1.0e9f;
      } else if (col >= nv) {
        s = -1.0e30f;                                   // pad column -> prob 0
      }
      ssc[row * 1024 + col] = s;
    }
  }
  __syncthreads();

  int r = tid >> 4, c16 = tid & 15;
  float m = -3.0e38f;
  for (int c = c16; c < 1024; c += 16) m = fmaxf(m, ssc[r * 1024 + c]);
  sred[r * 16 + c16] = m;
  __syncthreads();
  if (c16 == 0) {
    float mm = sred[r * 16];
    for (int j = 1; j < 16; ++j) mm = fmaxf(mm, sred[r * 16 + j]);
    rowstat[r] = mm;
  }
  __syncthreads();
  float rmax = rowstat[r];
  float ps = 0.f;
  for (int c = c16; c < 1024; c += 16) {
    float e = __expf(ssc[r * 1024 + c] - rmax);
    ssc[r * 1024 + c] = e;
    ps += e;
  }
  sred[r * 16 + c16] = ps;
  __syncthreads();
  if (c16 == 0) {
    float ss = 0.f;
    for (int j = 0; j < 16; ++j) ss += sred[r * 16 + j];
    rowstat[r] = 1.f / ss;
  }
  __syncthreads();
  float inv = rowstat[r];
  int q = q0 + r;
  if (q < nv) {
    float* orow = outp + ((size_t)bh * nv + q) * (size_t)nv;
    for (int c = c16; c < 1024; c += 16)
      if (c < nv) orow[c] = ssc[r * 1024 + c] * inv;
  }
}

// ---------------------------------------------------------------------------
// AV: hout[b][q][h*64+n] (=|+=) (a1(+a2)) @ V.  Probs read f32 from d_out,
// converted to bf16 on load.  rowshift=1 implements the pv_shift accumulation.
// ---------------------------------------------------------------------------
__global__ __launch_bounds__(128) void k_av(const float* __restrict__ a1,
                                            const float* __restrict__ a2, int nv,
                                            const unsigned short* __restrict__ Vt,
                                            float* __restrict__ hout, int rowshift) {
  int wave = threadIdx.x >> 5, lane = threadIdx.x & 31;
  int q0 = blockIdx.x * 16;
  int bh = blockIdx.y;
  int b = bh >> 3, h = bh & 7;
  int n0 = wave * 16;
  const float* aB  = a1 + (size_t)bh * nv * nv;
  const float* aB2 = a2 ? (a2 + (size_t)bh * nv * nv) : nullptr;
  const unsigned short* VtB = Vt + (size_t)bh * 65536;
  v8f acc = {};
  for (int kk = 0; kk < 1024; kk += 32) {
    bool fast = (q0 + 16 <= nv) && (kk + 32 <= nv);
    v16bf a  = load_frag_f32(aB, aB2, (size_t)nv, q0, kk, nv, fast);
    v16bf bf = load_frag_bf16(VtB, n0, 1024, kk);
    acc = wmma_bf16(a, bf, acc);
  }
  int col = h * 64 + n0 + (lane & 15);
  int mo = (lane < 16) ? 0 : 8;
#pragma unroll
  for (int i = 0; i < 8; ++i) {
    int q = q0 + i + mo;
    if (rowshift) {
      if (q < nv) hout[((size_t)(b * 1024 + q + 1)) * 512 + col] += acc[i];
    } else {
      hout[((size_t)(b * 1024 + q)) * 512 + col] = acc[i];
    }
  }
}

// ---------------------------------------------------------------------------
// Output FC: out = hin(f32->bf16) @ Bt + residual   (LN applied afterwards)
// ---------------------------------------------------------------------------
__global__ __launch_bounds__(128) void k_fc(const float* __restrict__ hin,
                                            const unsigned short* __restrict__ Bt,
                                            const float* __restrict__ resid,
                                            float* __restrict__ outp, int nrows) {
  int wave = threadIdx.x >> 5, lane = threadIdx.x & 31;
  int m0 = blockIdx.x * 16;
  int n0 = (blockIdx.y * 4 + wave) * 16;
  v8f acc = {};
  for (int kk = 0; kk < 512; kk += 32) {
    v16bf a = load_frag_f32(hin, nullptr, 512, m0, kk, 1 << 30, true);
    v16bf b = load_frag_bf16(Bt, n0, 512, kk);
    acc = wmma_bf16(a, b, acc);
  }
  int n = n0 + (lane & 15);
  int mo = (lane < 16) ? 0 : 8;
#pragma unroll
  for (int i = 0; i < 8; ++i) {
    int m = m0 + i + mo;
    int b_ = m >> 10, l = m & 1023;
    if (l < nrows) {
      size_t idx = ((size_t)(b_ * nrows + l)) * 512 + n;
      outp[idx] = acc[i] + resid[idx];
    }
  }
}

// ---------------------------------------------------------------------------
// In-place LayerNorm over 512 features.  One block per row.
// ---------------------------------------------------------------------------
__global__ __launch_bounds__(256) void k_ln(float* __restrict__ data,
                                            const float* __restrict__ g,
                                            const float* __restrict__ b) {
  __shared__ float s1[256], s2[256];
  int tid = threadIdx.x;
  float* p = data + (size_t)blockIdx.x * 512;
  float x0 = p[tid], x1 = p[tid + 256];
  s1[tid] = x0 + x1;
  s2[tid] = x0 * x0 + x1 * x1;
  __syncthreads();
  for (int s = 128; s > 0; s >>= 1) {
    if (tid < s) { s1[tid] += s1[tid + s]; s2[tid] += s2[tid + s]; }
    __syncthreads();
  }
  float mean = s1[0] * (1.f / 512.f);
  float var  = s2[0] * (1.f / 512.f) - mean * mean;
  float inv  = rsqrtf(var + 1e-6f);
  p[tid]       = (x0 - mean) * inv * g[tid]       + b[tid];
  p[tid + 256] = (x1 - mean) * inv * g[tid + 256] + b[tid + 256];
}

// ===========================================================================
extern "C" void kernel_launch(void* const* d_in, const int* in_sizes, int n_in,
                              void* d_out, int out_size, void* d_ws, size_t ws_size,
                              hipStream_t stream) {
  (void)in_sizes; (void)n_in; (void)out_size; (void)ws_size;
  const float* poi  = (const float*)d_in[0];
  const float* tiv  = (const float*)d_in[1];
  const float* dis  = (const float*)d_in[2];
  const int*   mask = (const int*)d_in[3];
  const float* g_q  = (const float*)d_in[16];
  const float* b_q  = (const float*)d_in[17];
  const float* g_iq = (const float*)d_in[18];
  const float* b_iq = (const float*)d_in[19];
  const float* g_dq = (const float*)d_in[20];
  const float* b_dq = (const float*)d_in[21];
  float* out = (float*)d_out;

  // d_out regions (flat, in return order)
  float* poi_out  = out;
  float* time_out = out + 2097152;
  float* dist_out = out + 4192256;
  float* attn1    = out + 6287360;
  float* attn2    = out + 39841792;
  float* attn3    = out + 73330720;

  // workspace layout (bytes)
  char* ws = (char*)d_ws;
  unsigned short* Wt  = (unsigned short*)ws;                    // 12x512x512 bf16
  unsigned short* Xp  = (unsigned short*)(ws + 6291456);        // 5x4096x512 bf16
  unsigned short* Qkv = (unsigned short*)(ws + 27262976);       // 9x32x1024x64 bf16
  unsigned short* Vt  = (unsigned short*)(ws + 65011712);       // 3x32x64x1024 bf16
  float*          Hout= (float*)(ws + 77594624);                // 3x4096x512 f32

  WPtrs wp;
  for (int i = 0; i < 12; ++i) wp.w[i] = (const float*)d_in[4 + i];

  k_prep_w<<<(12 * 262144 + 255) / 256, 256, 0, stream>>>(wp, Wt);
  k_prep_x<<<(5 * 2097152 + 255) / 256, 256, 0, stream>>>(poi, tiv, dis, Xp);

  // projections: {A-select, weight idx 0..8} -> q/k/v slot i
  const int xsel[9] = {0, 0, 0, 1, 1, 3, 2, 2, 4};
  for (int i = 0; i < 9; ++i) {
    k_proj<<<dim3(256, 8), 128, 0, stream>>>(
        Xp + (size_t)xsel[i] * 2097152, Wt + (size_t)i * 262144,
        Qkv + (size_t)i * 2097152);
  }

  k_transpose_v<<<dim3(64, 4, 96), 256, 0, stream>>>(Qkv, Vt);

  const size_t QK = 2097152;
  size_t attn_lds = (16 * 1024 + 256 + 16) * sizeof(float);
  k_attn<<<dim3(64, 32), 256, attn_lds, stream>>>(Qkv + 0 * QK, Qkv + 1 * QK, mask,    attn1, 1024);
  k_attn<<<dim3(64, 32), 256, attn_lds, stream>>>(Qkv + 3 * QK, Qkv + 4 * QK, nullptr, attn2, 1023);
  k_attn<<<dim3(64, 32), 256, attn_lds, stream>>>(Qkv + 6 * QK, Qkv + 7 * QK, nullptr, attn3, 1023);

  k_av<<<dim3(64, 32), 128, 0, stream>>>(attn1, nullptr, 1024, Vt + 0 * QK, Hout + 0 * QK, 0);
  k_av<<<dim3(64, 32), 128, 0, stream>>>(attn2, nullptr, 1023, Vt + 1 * QK, Hout + 1 * QK, 0);
  k_av<<<dim3(64, 32), 128, 0, stream>>>(attn3, nullptr, 1023, Vt + 2 * QK, Hout + 2 * QK, 0);
  k_av<<<dim3(64, 32), 128, 0, stream>>>(attn2, attn3,   1023, Vt + 0 * QK, Hout + 0 * QK, 1); // extra

  k_fc<<<dim3(256, 8), 128, 0, stream>>>(Hout + 0 * QK, Wt +  9 * 262144, poi, poi_out,  1024);
  k_fc<<<dim3(256, 8), 128, 0, stream>>>(Hout + 1 * QK, Wt + 10 * 262144, tiv, time_out, 1023);
  k_fc<<<dim3(256, 8), 128, 0, stream>>>(Hout + 2 * QK, Wt + 11 * 262144, dis, dist_out, 1023);

  k_ln<<<4096, 256, 0, stream>>>(poi_out,  g_q,  b_q);
  k_ln<<<4092, 256, 0, stream>>>(time_out, g_iq, b_iq);
  k_ln<<<4092, 256, 0, stream>>>(dist_out, g_dq, b_dq);
}
